// ResidualHead_49830210568639
// MI455X (gfx1250) — matrix operand order repaired and checked by reading, not compiled, not run on hardware
//
#include <hip/hip_runtime.h>
#include <hip/hip_bf16.h>
#include <math.h>

// ---------------------------------------------------------------------------
// ResidualHead fused pipeline for MI455X (gfx1250, wave32, WMMA)
//   B=64, N=1024, D=2048, H=1024
//   Main kernel: LN1(r) -> bf16 -> v_wmma_f32_16x16x32_bf16 GEMM vs packed W
//                -> LN2 in accumulators -> dot with q_proj -> scores
// ---------------------------------------------------------------------------

typedef __attribute__((ext_vector_type(16))) __bf16 v16bf;
typedef __attribute__((ext_vector_type(8)))  float  v8f;

#define EPS_LN   1e-5f
#define A_STRIDE 2056                      // halves per LDS A row (2048 + 8 pad, 16B aligned)
#define LDS_A_BYTES (32 * A_STRIDE * 2)    // 131584 B

__device__ __forceinline__ unsigned short f2bf(float x) {
    union { float f; unsigned u; } v; v.f = x;
    unsigned r = (v.u + 0x7FFFu + ((v.u >> 16) & 1u)) >> 16;   // RNE
    return (unsigned short)r;
}

__device__ __forceinline__ v8f zero8() {
    v8f z = {0.f,0.f,0.f,0.f,0.f,0.f,0.f,0.f};
    return z;
}

union FragU { uint4 q[2]; v16bf v; };

// ---------------------------------------------------------------------------
// Kernel 1: pack r_W [2048,1024] fp32 -> bf16 in WMMA B-fragment order.
// Packed layout (B, 32x16, contiguous k matching the A loads): lane L holds N = L%16, halves h -> K = (L/16)*16 + h.
// Linear index o = ((kt*64 + nt)*32 + lane)*16 + h.
// ---------------------------------------------------------------------------
__global__ __launch_bounds__(256) void pack_w_kernel(const float* __restrict__ rW,
                                                     unsigned short* __restrict__ Wp) {
    int o    = blockIdx.x * 256 + threadIdx.x;       // 0 .. 2M-1
    int h    = o & 15;
    int lane = (o >> 4) & 31;
    int tile = o >> 9;
    int nt   = tile & 63;
    int kt   = tile >> 6;
    int k    = kt * 32 + ((lane >> 4) << 4) + h;
    int n    = nt * 16 + (lane & 15);
    Wp[o] = f2bf(rW[(size_t)k * 1024 + n]);
}

// ---------------------------------------------------------------------------
// Kernel 2: q_proj = LN(LN(q) @ qW + qb).  64 blocks x 256 thr (tiny, fp32 VALU).
// ---------------------------------------------------------------------------
__global__ __launch_bounds__(256) void qproj_kernel(
    const float* __restrict__ q,  const float* __restrict__ g1, const float* __restrict__ b1,
    const float* __restrict__ qW, const float* __restrict__ qb,
    const float* __restrict__ g2, const float* __restrict__ b2,
    float* __restrict__ qproj) {
    __shared__ float xh[2048];
    __shared__ float red[256];
    const int tid = threadIdx.x, b = blockIdx.x;
    const float* qr = q + (size_t)b * 2048;

    float4 x0 = *(const float4*)(qr + tid * 4);
    float4 x1 = *(const float4*)(qr + 1024 + tid * 4);
    float s  = x0.x + x0.y + x0.z + x0.w + x1.x + x1.y + x1.z + x1.w;
    float ss = x0.x*x0.x + x0.y*x0.y + x0.z*x0.z + x0.w*x0.w
             + x1.x*x1.x + x1.y*x1.y + x1.z*x1.z + x1.w*x1.w;

    red[tid] = s;  __syncthreads();
    for (int st = 128; st > 0; st >>= 1) { if (tid < st) red[tid] += red[tid + st]; __syncthreads(); }
    float tot = red[0]; __syncthreads();
    red[tid] = ss; __syncthreads();
    for (int st = 128; st > 0; st >>= 1) { if (tid < st) red[tid] += red[tid + st]; __syncthreads(); }
    float tot2 = red[0]; __syncthreads();

    float mu = tot * (1.f/2048.f);
    float rs = rsqrtf(tot2 * (1.f/2048.f) - mu*mu + EPS_LN);
    {
        int d = tid * 4;
        float4 gg = *(const float4*)(g1 + d), bb = *(const float4*)(b1 + d);
        xh[d+0] = (x0.x-mu)*rs*gg.x + bb.x;  xh[d+1] = (x0.y-mu)*rs*gg.y + bb.y;
        xh[d+2] = (x0.z-mu)*rs*gg.z + bb.z;  xh[d+3] = (x0.w-mu)*rs*gg.w + bb.w;
        d += 1024;
        gg = *(const float4*)(g1 + d); bb = *(const float4*)(b1 + d);
        xh[d+0] = (x1.x-mu)*rs*gg.x + bb.x;  xh[d+1] = (x1.y-mu)*rs*gg.y + bb.y;
        xh[d+2] = (x1.z-mu)*rs*gg.z + bb.z;  xh[d+3] = (x1.w-mu)*rs*gg.w + bb.w;
    }
    __syncthreads();

    const int h0 = tid * 4;
    float4 acc = {0.f, 0.f, 0.f, 0.f};
    for (int d = 0; d < 2048; ++d) {
        float xv = xh[d];                                    // LDS broadcast
        float4 w = *(const float4*)(qW + (size_t)d * 1024 + h0);
        acc.x += xv * w.x; acc.y += xv * w.y; acc.z += xv * w.z; acc.w += xv * w.w;
    }
    float4 bq = *(const float4*)(qb + h0);
    acc.x += bq.x; acc.y += bq.y; acc.z += bq.z; acc.w += bq.w;

    float s2 = acc.x + acc.y + acc.z + acc.w;
    float q2 = acc.x*acc.x + acc.y*acc.y + acc.z*acc.z + acc.w*acc.w;
    red[tid] = s2; __syncthreads();
    for (int st = 128; st > 0; st >>= 1) { if (tid < st) red[tid] += red[tid + st]; __syncthreads(); }
    float t1 = red[0]; __syncthreads();
    red[tid] = q2; __syncthreads();
    for (int st = 128; st > 0; st >>= 1) { if (tid < st) red[tid] += red[tid + st]; __syncthreads(); }
    float t2 = red[0]; __syncthreads();

    float mu2 = t1 * (1.f/1024.f);
    float rs2 = rsqrtf(t2 * (1.f/1024.f) - mu2*mu2 + EPS_LN);
    float4 gg2 = *(const float4*)(g2 + h0), bb2 = *(const float4*)(b2 + h0);
    float* o = qproj + (size_t)b * 1024 + h0;
    o[0] = (acc.x-mu2)*rs2*gg2.x + bb2.x;  o[1] = (acc.y-mu2)*rs2*gg2.y + bb2.y;
    o[2] = (acc.z-mu2)*rs2*gg2.z + bb2.z;  o[3] = (acc.w-mu2)*rs2*gg2.w + bb2.w;
}

// ---------------------------------------------------------------------------
// Kernel 3: main fused kernel.  grid = (32 row-blocks, 64 batch), 512 thr = 16 waves.
// Block owns 32 rows x full H=1024.  Each wave: 64 H-cols (4 n-tiles) x 2 M-tiles.
// ---------------------------------------------------------------------------
__global__ __launch_bounds__(512) void rproj_score_kernel(
    const float* __restrict__ r,  const float* __restrict__ g1, const float* __restrict__ b1,
    const unsigned short* __restrict__ Wp,
    const float* __restrict__ rb, const float* __restrict__ g2, const float* __restrict__ b2,
    const float* __restrict__ qproj, float* __restrict__ scores) {
    extern __shared__ char smem[];
    unsigned short* Alds = (unsigned short*)smem;
    float* red1 = (float*)(smem + LDS_A_BYTES);
    float* red2 = red1 + 512;
    float* muL  = red2 + 512;
    float* rsL  = muL + 32;

    const int tid = threadIdx.x, wid = tid >> 5, lane = tid & 31;
    const int g = lane >> 4, ml = lane & 15;
    const int b  = blockIdx.y;
    const int n0 = blockIdx.x * 32;

    // ---- Phase 1: LN1 of 32 rows of r -> bf16 A-tiles in LDS (1 wave per 2 rows)
    for (int rr = 0; rr < 2; ++rr) {
        const int row = wid * 2 + rr;
        const float* rp = r + ((size_t)(b * 1024 + n0 + row)) * 2048;
        float4 x[16]; float s = 0.f, ss = 0.f;
        #pragma unroll
        for (int i = 0; i < 16; ++i) {
            x[i] = *(const float4*)(rp + i * 128 + lane * 4);   // 512B/ wave / iter, coalesced
            s  += x[i].x + x[i].y + x[i].z + x[i].w;
            ss += x[i].x*x[i].x + x[i].y*x[i].y + x[i].z*x[i].z + x[i].w*x[i].w;
        }
        #pragma unroll
        for (int off = 16; off > 0; off >>= 1) {
            s  += __shfl_xor(s,  off, 32);
            ss += __shfl_xor(ss, off, 32);
        }
        const float mu = s * (1.f/2048.f);
        const float rs = rsqrtf(ss * (1.f/2048.f) - mu*mu + EPS_LN);
        #pragma unroll
        for (int i = 0; i < 16; ++i) {
            const int d = i * 128 + lane * 4;
            float4 gg = *(const float4*)(g1 + d), bb = *(const float4*)(b1 + d);
            unsigned short h0 = f2bf((x[i].x-mu)*rs*gg.x + bb.x);
            unsigned short h1 = f2bf((x[i].y-mu)*rs*gg.y + bb.y);
            unsigned short h2 = f2bf((x[i].z-mu)*rs*gg.z + bb.z);
            unsigned short h3 = f2bf((x[i].w-mu)*rs*gg.w + bb.w);
            uint2 pk; pk.x = (unsigned)h0 | ((unsigned)h1 << 16);
                      pk.y = (unsigned)h2 | ((unsigned)h3 << 16);
            *(uint2*)(Alds + row * A_STRIDE + d) = pk;
        }
    }
    __syncthreads();

    // ---- Phase 2: WMMA GEMM.  acc[mt][t] = 16x16 f32 tile.
    v8f acc[2][4];
    #pragma unroll
    for (int mt = 0; mt < 2; ++mt)
        #pragma unroll
        for (int t = 0; t < 4; ++t) acc[mt][t] = zero8();

    for (int kt = 0; kt < 64; ++kt) {
        // A fragments (contiguous k order, same as packed B: K = g*16+{0..15})
        FragU a0, a1;
        {
            const int base0 = ml * A_STRIDE + kt * 32 + g * 16;
            a0.q[0] = *(const uint4*)(Alds + base0);
            a0.q[1] = *(const uint4*)(Alds + base0 + 8);
            const int base1 = (16 + ml) * A_STRIDE + kt * 32 + g * 16;
            a1.q[0] = *(const uint4*)(Alds + base1);
            a1.q[1] = *(const uint4*)(Alds + base1 + 8);
        }
        // Prefetch next kt's first B fragment (speculative, dropped if OOB)
        __builtin_prefetch(Wp + (((size_t)(kt + 1) * 64 + wid * 4) * 32 + lane) * 16, 0, 0);
        #pragma unroll
        for (int t = 0; t < 4; ++t) {
            const int nt = wid * 4 + t;
            const unsigned short* bp = Wp + (((size_t)kt * 64 + nt) * 32 + lane) * 16;
            FragU bu;
            bu.q[0] = *(const uint4*)(bp);       // halves 0..7   (global_load_b128)
            bu.q[1] = *(const uint4*)(bp + 8);   // halves 8..15
            acc[0][t] = __builtin_amdgcn_wmma_f32_16x16x32_bf16(
                false, a0.v, false, bu.v, (short)0, acc[0][t], false, false);
            acc[1][t] = __builtin_amdgcn_wmma_f32_16x16x32_bf16(
                false, a1.v, false, bu.v, (short)0, acc[1][t], false, false);
        }
    }

    // ---- Phase 3: epilogue.  C layout: VGPR j, lanes 0-15 -> M=j, lanes 16-31 -> M=j+8, N=lane%16.
    float rbv[4], g2v[4], b2v[4], qpv[4];
    #pragma unroll
    for (int t = 0; t < 4; ++t) {
        const int h = wid * 64 + t * 16 + ml;
        rbv[t] = rb[h]; g2v[t] = g2[h]; b2v[t] = b2[h];
        qpv[t] = qproj[(size_t)b * 1024 + h];
    }

    float s1[2][8], s2[2][8];
    #pragma unroll
    for (int mt = 0; mt < 2; ++mt)
        #pragma unroll
        for (int j = 0; j < 8; ++j) { s1[mt][j] = 0.f; s2[mt][j] = 0.f; }
    #pragma unroll
    for (int mt = 0; mt < 2; ++mt)
        #pragma unroll
        for (int t = 0; t < 4; ++t)
            #pragma unroll
            for (int j = 0; j < 8; ++j) {
                float y = acc[mt][t][j] + rbv[t];
                s1[mt][j] += y; s2[mt][j] += y * y;
            }
    #pragma unroll
    for (int off = 1; off < 16; off <<= 1)
        #pragma unroll
        for (int mt = 0; mt < 2; ++mt)
            #pragma unroll
            for (int j = 0; j < 8; ++j) {
                s1[mt][j] += __shfl_xor(s1[mt][j], off, 32);
                s2[mt][j] += __shfl_xor(s2[mt][j], off, 32);
            }
    if (ml == 0) {
        #pragma unroll
        for (int mt = 0; mt < 2; ++mt)
            #pragma unroll
            for (int j = 0; j < 8; ++j) {
                const int row = mt * 16 + g * 8 + j;
                red1[row * 16 + wid] = s1[mt][j];
                red2[row * 16 + wid] = s2[mt][j];
            }
    }
    __syncthreads();
    if (tid < 32) {
        float a = 0.f, c = 0.f;
        for (int w = 0; w < 16; ++w) { a += red1[tid * 16 + w]; c += red2[tid * 16 + w]; }
        const float m = a * (1.f/1024.f);
        muL[tid] = m;
        rsL[tid] = rsqrtf(c * (1.f/1024.f) - m * m + EPS_LN);
    }
    __syncthreads();

    float p[2][8];
    #pragma unroll
    for (int mt = 0; mt < 2; ++mt)
        #pragma unroll
        for (int j = 0; j < 8; ++j) {
            const int row = mt * 16 + g * 8 + j;
            const float m = muL[row], rs = rsL[row];
            float acc_p = 0.f;
            #pragma unroll
            for (int t = 0; t < 4; ++t) {
                float y = acc[mt][t][j] + rbv[t];
                acc_p += ((y - m) * rs * g2v[t] + b2v[t]) * qpv[t];
            }
            p[mt][j] = acc_p;
        }
    #pragma unroll
    for (int off = 1; off < 16; off <<= 1)
        #pragma unroll
        for (int mt = 0; mt < 2; ++mt)
            #pragma unroll
            for (int j = 0; j < 8; ++j)
                p[mt][j] += __shfl_xor(p[mt][j], off, 32);
    if (ml == 0) {
        #pragma unroll
        for (int mt = 0; mt < 2; ++mt)
            #pragma unroll
            for (int j = 0; j < 8; ++j) {
                const int row = mt * 16 + g * 8 + j;
                red1[row * 16 + wid] = p[mt][j];
            }
    }
    __syncthreads();
    if (tid < 32) {
        float sc = 0.f;
        for (int w = 0; w < 16; ++w) sc += red1[tid * 16 + w];
        scores[(size_t)b * 1024 + n0 + tid] = sc * 0.03125f;   // / sqrt(1024)
    }
}

// ---------------------------------------------------------------------------
// Kernel 4: per-batch top-32 -> softmax -> weighted ref_vals -> Platt sigmoid.
// 32x argmax gives exact top-k semantics of the masked softmax.
// ---------------------------------------------------------------------------
__global__ __launch_bounds__(256) void topk_softmax_platt_kernel(
    const float* __restrict__ scores, const float* __restrict__ ref_vals,
    const float* __restrict__ scale,  const float* __restrict__ res_scale,
    const int* __restrict__ tau, float* __restrict__ out) {
    __shared__ float sc[1024];
    __shared__ float rmx[256];
    __shared__ int   ridx[256];
    const int tid = threadIdx.x, b = blockIdx.x;
    const float* srow = scores + (size_t)b * 1024;
    #pragma unroll
    for (int i = 0; i < 4; ++i) sc[tid * 4 + i] = srow[tid * 4 + i];
    __syncthreads();

    const float invTau = 1.0f / (float)(*tau);
    float num = 0.f, den = 0.f, m0 = 0.f;          // meaningful only on tid 0
    for (int it = 0; it < 32; ++it) {
        float mv = -3e38f; int mi = 0;
        #pragma unroll
        for (int i = 0; i < 4; ++i) {
            float v = sc[tid * 4 + i];
            if (v > mv) { mv = v; mi = tid * 4 + i; }
        }
        rmx[tid] = mv; ridx[tid] = mi;
        __syncthreads();
        for (int st = 128; st > 0; st >>= 1) {
            if (tid < st && rmx[tid + st] > rmx[tid]) { rmx[tid] = rmx[tid + st]; ridx[tid] = ridx[tid + st]; }
            __syncthreads();
        }
        if (tid == 0) {
            const float mval = rmx[0]; const int idx = ridx[0];
            if (it == 0) m0 = mval;
            const float w = __expf((mval - m0) * invTau);
            num += w * ref_vals[(size_t)b * 1024 + idx];
            den += w;
            sc[idx] = -3e38f;
        }
        __syncthreads();
    }
    if (tid == 0) {
        float pv = num / den;
        pv = fminf(fmaxf(pv, 1e-4f), 1.f - 1e-4f);
        const float logit = logf(pv) - log1pf(-pv);
        const float z = scale[0] * logit + res_scale[0];
        out[b] = 1.f / (1.f + __expf(-z));
    }
}

// ---------------------------------------------------------------------------
extern "C" void kernel_launch(void* const* d_in, const int* in_sizes, int n_in,
                              void* d_out, int out_size, void* d_ws, size_t ws_size,
                              hipStream_t stream) {
    const float* q        = (const float*)d_in[0];
    const float* r        = (const float*)d_in[1];
    const float* ref_vals = (const float*)d_in[2];
    const float* q_ln1_g  = (const float*)d_in[3];
    const float* q_ln1_b  = (const float*)d_in[4];
    const float* q_W      = (const float*)d_in[5];
    const float* q_b      = (const float*)d_in[6];
    const float* q_ln2_g  = (const float*)d_in[7];
    const float* q_ln2_b  = (const float*)d_in[8];
    const float* r_ln1_g  = (const float*)d_in[9];
    const float* r_ln1_b  = (const float*)d_in[10];
    const float* r_W      = (const float*)d_in[11];
    const float* r_b      = (const float*)d_in[12];
    const float* r_ln2_g  = (const float*)d_in[13];
    const float* r_ln2_b  = (const float*)d_in[14];
    const float* scale    = (const float*)d_in[15];
    const float* res_sc   = (const float*)d_in[16];
    const int*   tau      = (const int*)d_in[17];
    float* out = (float*)d_out;

    unsigned short* Wp   = (unsigned short*)d_ws;                         // 4 MB bf16 packed W
    float* qproj  = (float*)((char*)d_ws + (size_t)4 * 1024 * 1024);      // 256 KB
    float* scores = qproj + 64 * 1024;                                    // 256 KB

    const size_t smemR = LDS_A_BYTES + 2 * 512 * sizeof(float) + 2 * 32 * sizeof(float);
    static bool attr_set = []{
        hipFuncSetAttribute((const void*)rproj_score_kernel,
                            hipFuncAttributeMaxDynamicSharedMemorySize, (int)(LDS_A_BYTES + 8192));
        return true;
    }();
    (void)attr_set;

    pack_w_kernel<<<8192, 256, 0, stream>>>(r_W, Wp);
    qproj_kernel<<<64, 256, 0, stream>>>(q, q_ln1_g, q_ln1_b, q_W, q_b, q_ln2_g, q_ln2_b, qproj);
    dim3 gridR(32, 64);
    rproj_score_kernel<<<gridR, 512, smemR, stream>>>(r, r_ln1_g, r_ln1_b, Wp,
                                                      r_b, r_ln2_g, r_ln2_b, qproj, scores);
    topk_softmax_platt_kernel<<<64, 256, 0, stream>>>(scores, ref_vals, scale, res_sc, tau, out);
}
